// MolecularGNN_5059471475382
// MI455X (gfx1250) — matrix operand order repaired and checked by reading, not compiled, not run on hardware
//
#include <hip/hip_runtime.h>

// CDNA5 / gfx1250 bf16 WMMA GEMM pipeline for the 3-layer GCN.
// v3: 4x4 register blocking (64x64/wave) + block-shared LDS staging of the
// A macro-tile (8 waves per block share one macro-row), double-buffered with
// one barrier per k-step. A-fragments come from ds_load_b128 (padded stride,
// conflict-free); B-fragments stay as contiguous 32B global loads.

typedef __attribute__((ext_vector_type(16))) __bf16 v16bf;
typedef __attribute__((ext_vector_type(8)))  __bf16 v8bf;
typedef __attribute__((ext_vector_type(8)))  float  v8f;

#define BN_EPS 1e-5f
#define LDK 40   // padded LDS row stride (elements): 80B -> 16 distinct banks

// ---------------- elementwise f32 -> bf16 convert ----------------
__global__ void cvt_f32_bf16(const float* __restrict__ in,
                             __bf16* __restrict__ out, long n) {
  long i = (long)blockIdx.x * blockDim.x + threadIdx.x;
  long stride = (long)gridDim.x * blockDim.x;
  for (; i < n; i += stride) out[i] = (__bf16)in[i];
}

// -------- transpose + convert: in [K,N] f32 row-major -> out [N,K] bf16 ----
__global__ void cvt_t_f32_bf16(const float* __restrict__ in,
                               __bf16* __restrict__ out, int K, int N) {
  int idx = blockIdx.x * blockDim.x + threadIdx.x;
  if (idx >= K * N) return;
  int k = idx / N;
  int n = idx - k * N;
  out[(long)n * K + k] = (__bf16)in[idx];
}

// ---------------- generic batched bf16 WMMA GEMM ----------------
// C[M,N] = A[M,K] @ Bt[N,K]^T   (Bt is B stored N-major, K contiguous)
// One wave computes a 64x64 macro-tile = 4x4 grid of 16x16 WMMA tiles.
// USE_LDS=1 (requires N==512, i.e. 8 macro-cols): the 8 waves of a block
//   cover one macro-row and share the 64x32 A tile through LDS.
// USE_LDS=0: fully independent waves, direct global fragment loads.
// Grids sized to exact macro-tile counts -> EXEC all-1s (WMMA requirement).
template <int USE_LDS>
__global__ __launch_bounds__(256) void gemm_wmma_bf16(
    const __bf16* __restrict__ A, const __bf16* __restrict__ Bt,
    float* __restrict__ Cf, __bf16* __restrict__ Cb,
    const float* __restrict__ bias,
    const float* __restrict__ gamma, const float* __restrict__ beta,
    const float* __restrict__ rmean, const float* __restrict__ rvar,
    int M, int N, int K,
    long strideA, long strideB, long strideC,
    int ldc, int storeT) {
  extern __shared__ __bf16 smA[];  // [2][64][LDK] when USE_LDS

  const int wave = threadIdx.x >> 5;
  const int lane = threadIdx.x & 31;
  const int lo = lane & 15;   // row within tile (A) / col within tile (B,C)
  const int hi = lane >> 4;   // half-wave selector

  int tm, tn;
  if (USE_LDS) {
    tm = blockIdx.x;          // one macro-row per block
    tn = wave;                // 8 waves cover N=512
  } else {
    const int tilesN = N >> 6;
    const int tileId = blockIdx.x * 8 + wave;
    tm = tileId / tilesN;
    tn = tileId - tm * tilesN;
  }

  const long batch = blockIdx.y;
  const __bf16* pB = Bt + batch * strideB + (long)(tn * 64 + lo) * K;
  const long rowStrideB = (long)16 * K;

  v8f acc[4][4];
#pragma unroll
  for (int m = 0; m < 4; ++m)
#pragma unroll
    for (int n = 0; n < 4; ++n) acc[m][n] = (v8f){};

  if (USE_LDS) {
    // ---- LDS-staged A path ----
    const int t = threadIdx.x;
    const int sRow = t >> 2;   // 0..63
    const int sChk = t & 3;    // 0..3 (16B chunk within 64B row)
    const __bf16* gA =
        A + batch * strideA + (long)(tm * 64 + sRow) * K + sChk * 8;

    // prologue: stage k-tile 0 into buffer 0
    {
      v8bf s0 = *(const v8bf*)(gA);
      *(v8bf*)(&smA[sRow * LDK + sChk * 8]) = s0;
    }
    __syncthreads();

    const int KT = K >> 5;
#pragma unroll 1
    for (int kt = 0; kt < KT; ++kt) {
      const int buf = kt & 1;
      const __bf16* smBase = smA + buf * 64 * LDK;

      // issue next-tile global load early (overlaps with compute)
      v8bf nxt;
      const bool hasNext = (kt + 1 < KT);
      if (hasNext) nxt = *(const v8bf*)(gA + (kt + 1) * 32);

      v16bf aF[4], bF[4];
#pragma unroll
      for (int m = 0; m < 4; ++m) {
        // A fragment from LDS: row M=lo of subtile m,
        // chunks [hi*16,+8) and [hi*16+8,+8)  (two ds_load_b128)
        const __bf16* ps = smBase + (m * 16 + lo) * LDK;
        v8bf a0 = *(const v8bf*)(ps + hi * 16);
        v8bf a1 = *(const v8bf*)(ps + hi * 16 + 8);
#pragma unroll
        for (int i = 0; i < 8; ++i) { aF[m][i] = a0[i]; aF[m][i + 8] = a1[i]; }
      }
#pragma unroll
      for (int n = 0; n < 4; ++n)
        bF[n] = *(const v16bf*)(pB + n * rowStrideB + kt * 32 + hi * 16);
#pragma unroll
      for (int m = 0; m < 4; ++m)
#pragma unroll
        for (int n = 0; n < 4; ++n)
          acc[m][n] = __builtin_amdgcn_wmma_f32_16x16x32_bf16(
              false, aF[m], false, bF[n], (short)0, acc[m][n], false, false);

      if (hasNext) {
        // write other buffer; single barrier separates iterations
        *(v8bf*)(&smA[(buf ^ 1) * 64 * LDK + sRow * LDK + sChk * 8]) = nxt;
      }
      __syncthreads();
    }
  } else {
    // ---- direct-load path ----
    const __bf16* pA = A + batch * strideA + (long)(tm * 64 + lo) * K;
    const long rowStrideA = (long)16 * K;
#pragma unroll 1
    for (int k = 0; k < K; k += 32) {
      v16bf aF[4], bF[4];
#pragma unroll
      for (int m = 0; m < 4; ++m) {
        const __bf16* p = pA + m * rowStrideA + k;
        v8bf a0 = *(const v8bf*)(p + hi * 16);
        v8bf a1 = *(const v8bf*)(p + hi * 16 + 8);
#pragma unroll
        for (int i = 0; i < 8; ++i) { aF[m][i] = a0[i]; aF[m][i + 8] = a1[i]; }
      }
#pragma unroll
      for (int n = 0; n < 4; ++n)
        bF[n] = *(const v16bf*)(pB + n * rowStrideB + k + hi * 16);
#pragma unroll
      for (int m = 0; m < 4; ++m)
#pragma unroll
        for (int n = 0; n < 4; ++n)
          acc[m][n] = __builtin_amdgcn_wmma_f32_16x16x32_bf16(
              false, aF[m], false, bF[n], (short)0, acc[m][n], false, false);
    }
  }

  // epilogue: D layout -> VGPR r: M = r + 8*hi, N = lo (within 16x16 tile)
  const bool doBN = (gamma != nullptr);
#pragma unroll
  for (int n = 0; n < 4; ++n) {
    const int nCol = tn * 64 + n * 16 + lo;
    const float bv = bias ? bias[nCol] : 0.0f;
    float scale = 1.0f, shift = 0.0f;
    if (doBN) {
      float s = gamma[nCol] * rsqrtf(rvar[nCol] + BN_EPS);
      scale = s;
      shift = beta[nCol] - rmean[nCol] * s;
    }
#pragma unroll
    for (int m = 0; m < 4; ++m) {
#pragma unroll
      for (int r = 0; r < 8; ++r) {
        const int mRow = tm * 64 + m * 16 + r + hi * 8;
        float v = acc[m][n][r] + bv;
        if (doBN) v = fmaxf(v, 0.0f) * scale + shift;  // relu then BN (eval)
        const long off = batch * strideC +
                         (storeT ? ((long)nCol * ldc + mRow)
                                 : ((long)mRow * ldc + nCol));
        if (Cf) Cf[off] = v;
        else    Cb[off] = (__bf16)v;
      }
    }
  }
}

extern "C" void kernel_launch(void* const* d_in, const int* in_sizes, int n_in,
                              void* d_out, int out_size, void* d_ws, size_t ws_size,
                              hipStream_t stream) {
  const float* x     = (const float*)d_in[0];
  const float* adj   = (const float*)d_in[1];
  const float* W_in  = (const float*)d_in[2];
  const float* b_in  = (const float*)d_in[3];
  const float* Wg    = (const float*)d_in[4];
  const float* bg    = (const float*)d_in[5];
  const float* gamma = (const float*)d_in[6];
  const float* beta  = (const float*)d_in[7];
  const float* rmean = (const float*)d_in[8];
  const float* rvar  = (const float*)d_in[9];
  const float* W_out = (const float*)d_in[10];
  const float* b_out = (const float*)d_in[11];
  float* outp = (float*)d_out;
  (void)in_sizes; (void)n_in; (void)out_size; (void)ws_size;

  const int B = 32, Nn = 1024, F = 128, H = 512, O = 256;
  const long M = (long)B * Nn;  // 32768
  const size_t smBytes = (size_t)2 * 64 * LDK * 2;  // double-buffered A tile

  char* ws = (char*)d_ws;
  size_t off = 0;
  auto alloc = [&](size_t bytes) -> void* {
    void* p = ws + off;
    off += (bytes + 255) & ~((size_t)255);
    return p;
  };
  __bf16* xb    = (__bf16*)alloc((size_t)M * F * 2);          // x in bf16
  __bf16* adjb  = (__bf16*)alloc((size_t)B * Nn * Nn * 2);    // adj bf16 (fits L2)
  __bf16* hb    = (__bf16*)alloc((size_t)M * H * 2);          // activations
  __bf16* stb   = (__bf16*)alloc((size_t)B * H * Nn * 2);     // support^T per batch
  __bf16* WtIn  = (__bf16*)alloc((size_t)H * F * 2);          // W_in^T
  __bf16* WtG   = (__bf16*)alloc((size_t)3 * H * H * 2);      // Wg^T x3
  __bf16* WtOut = (__bf16*)alloc((size_t)O * H * 2);          // W_out^T

  // ---- precision conversion / weight transposition ----
  cvt_f32_bf16<<<2048, 256, 0, stream>>>(x, xb, M * F);
  cvt_f32_bf16<<<8192, 256, 0, stream>>>(adj, adjb, (long)B * Nn * Nn);
  cvt_t_f32_bf16<<<(F * H + 255) / 256, 256, 0, stream>>>(W_in, WtIn, F, H);
  for (int i = 0; i < 3; ++i)
    cvt_t_f32_bf16<<<(H * H + 255) / 256, 256, 0, stream>>>(
        Wg + (long)i * H * H, WtG + (long)i * H * H, H, H);
  cvt_t_f32_bf16<<<(H * O + 255) / 256, 256, 0, stream>>>(W_out, WtOut, H, O);

  // ---- input projection: hb = x @ W_in + b_in   [M x H]  (N=512 -> LDS) ----
  gemm_wmma_bf16<1><<<dim3((int)(M / 64), 1), 256, smBytes, stream>>>(
      xb, WtIn, nullptr, hb, b_in,
      nullptr, nullptr, nullptr, nullptr,
      (int)M, H, F, 0, 0, 0, H, /*storeT=*/0);

  // ---- 3 GCN layers (both GEMMs have N=512 -> LDS path) ----
  for (int i = 0; i < 3; ++i) {
    // support^T = (h @ Wg_i)^T  per batch: stb[b][H][Nn]
    gemm_wmma_bf16<1><<<dim3(Nn / 64, B), 256, smBytes, stream>>>(
        hb, WtG + (long)i * H * H, nullptr, stb, nullptr,
        nullptr, nullptr, nullptr, nullptr,
        Nn, H, H, (long)Nn * H, 0, (long)H * Nn, Nn, /*storeT=*/1);
    // h = BN(relu(adj @ support + bg_i))  per batch
    gemm_wmma_bf16<1><<<dim3(Nn / 64, B), 256, smBytes, stream>>>(
        adjb, stb, nullptr, hb, bg + (long)i * H,
        gamma + (long)i * H, beta + (long)i * H,
        rmean + (long)i * H, rvar + (long)i * H,
        Nn, H, Nn, (long)Nn * Nn, (long)H * Nn, (long)Nn * H, H, /*storeT=*/0);
  }

  // ---- output projection: d_out = h @ W_out + b_out  [M x O], f32 ----
  // N=256 (4 macro-cols) -> direct-load variant
  {
    int tiles = (int)(M / 64) * (O / 64);  // 512*4 = 2048 macro-tiles
    gemm_wmma_bf16<0><<<dim3(tiles / 8, 1), 256, 0, stream>>>(
        hb, WtOut, outp, nullptr, b_out,
        nullptr, nullptr, nullptr, nullptr,
        (int)M, O, H, 0, 0, 0, O, /*storeT=*/0);
  }
}
